// QuantumCircuitOptimizer_27925877358871
// MI455X (gfx1250) — matrix operand order repaired and checked by reading, not compiled, not run on hardware
//
#include <hip/hip_runtime.h>

// ---------------- CDNA5 feature probes (compile-safe) ----------------
#if defined(__has_builtin)
#  if __has_builtin(__builtin_amdgcn_global_load_async_to_lds_b128)
#    define QC_ASYNC 1
#  endif
#endif
#ifndef QC_ASYNC
#  define QC_ASYNC 0
#endif

namespace {

constexpr int NLAYERS = 16;
constexpr int QDIM    = 2048;
constexpr int BATCH   = 4096;

constexpr int BM = 128, BN = 64, BK = 32;
constexpr int KTILES = QDIM / BK;     // 64
constexpr int LDSK   = BK + 8;        // 40 bf16 = 80B row stride (16B aligned, bank-spread)

typedef __attribute__((ext_vector_type(16))) __bf16 v16bf;
typedef __attribute__((ext_vector_type(8)))  __bf16 v8bf;
typedef __attribute__((ext_vector_type(8)))  float  v8f;
typedef __attribute__((ext_vector_type(4)))  int    v4i;

__device__ __forceinline__ void split_bf16(float x, __bf16& hi, __bf16& lo) {
  hi = static_cast<__bf16>(x);
  lo = static_cast<__bf16>(x - static_cast<float>(hi));
}

__device__ __forceinline__ void qc_wait_async() {
#if QC_ASYNC
#  if __has_builtin(__builtin_amdgcn_s_wait_asynccnt)
  __builtin_amdgcn_s_wait_asynccnt(0);
#  else
  asm volatile("s_wait_asynccnt 0" ::: "memory");
#  endif
#endif
}

#if QC_ASYNC
__device__ __forceinline__ void copy16_async(const __bf16* g, __bf16* l) {
  // signature (from hipcc diagnostic): (v4i AS(1)*, v4i AS(3)*, int offset, int cpol)
  __builtin_amdgcn_global_load_async_to_lds_b128(
      (__attribute__((address_space(1))) v4i*)(v4i*)g,
      (__attribute__((address_space(3))) v4i*)(v4i*)l, 0, 0);
}
#endif

} // namespace

// scales[l][d] = prod_j cos(0.5 * angles[l][d][j])
__global__ void qc_scales_kernel(const float* __restrict__ angles,
                                 float* __restrict__ scales, int n) {
  int i = blockIdx.x * blockDim.x + threadIdx.x;
  if (i < n) {
    float a0 = angles[3 * i + 0];
    float a1 = angles[3 * i + 1];
    float a2 = angles[3 * i + 2];
    scales[i] = __cosf(0.5f * a0) * __cosf(0.5f * a1) * __cosf(0.5f * a2);
  }
}

// Split input state (with layer-0 scale folded in) into bf16 hi/lo.
__global__ void qc_split_in(const float* __restrict__ X,
                            const float* __restrict__ scale0,
                            __bf16* __restrict__ hi, __bf16* __restrict__ lo) {
  size_t i = (size_t)blockIdx.x * blockDim.x + threadIdx.x;
  if (i < (size_t)BATCH * QDIM) {
    float v = X[i] * scale0[i & (QDIM - 1)];
    __bf16 h, l;
    split_bf16(v, h, l);
    hi[i] = h;
    lo[i] = l;
  }
}

// Transpose + split one weight matrix: W[k][n] fp32 -> Wt_hi/lo[n][k] bf16.
__global__ __launch_bounds__(256)
void qc_split_w(const float* __restrict__ W,
                __bf16* __restrict__ Wthi, __bf16* __restrict__ Wtlo) {
  __shared__ float t[32][33];
  const int tx = threadIdx.x & 31;
  const int ty = threadIdx.x >> 5;              // 0..7
  const int n0 = blockIdx.x * 32;
  const int k0 = blockIdx.y * 32;
#pragma unroll
  for (int j = 0; j < 4; ++j)
    t[ty + j * 8][tx] = W[(size_t)(k0 + ty + j * 8) * QDIM + n0 + tx];
  __syncthreads();
#pragma unroll
  for (int j = 0; j < 4; ++j) {
    const int nl = ty + j * 8;
    float v = t[tx][nl];                         // = W[k0+tx][n0+nl]
    __bf16 h, l;
    split_bf16(v, h, l);
    Wthi[(size_t)(n0 + nl) * QDIM + k0 + tx] = h;
    Wtlo[(size_t)(n0 + nl) * QDIM + k0 + tx] = l;
  }
}

// C = A @ W using bf16 hi/lo split operands, fp32 accumulate:
//   AB ~= Ahi*Bhi + Ahi*Blo + Alo*Bhi
// Epilogue: if Cf32 -> plain fp32 store; else split(acc * nextScale[n]) -> Chi/Clo.
__global__ __launch_bounds__(256)
void qc_gemm_bf16(const __bf16* __restrict__ Ahi, const __bf16* __restrict__ Alo,
                  const __bf16* __restrict__ Wthi, const __bf16* __restrict__ Wtlo,
                  const float* __restrict__ nextScale,   // may be nullptr
                  __bf16* __restrict__ Chi, __bf16* __restrict__ Clo,
                  float* __restrict__ Cf32) {            // may be nullptr
  __shared__ __bf16 sAhi[2][BM * LDSK];   // 20480 B
  __shared__ __bf16 sAlo[2][BM * LDSK];   // 20480 B
  __shared__ __bf16 sWhi[2][BN * LDSK];   // 10240 B
  __shared__ __bf16 sWlo[2][BN * LDSK];   // 10240 B  -> total 61440 B

  const int tid = threadIdx.x;
  const int m0  = blockIdx.y * BM;
  const int n0  = blockIdx.x * BN;

#if !QC_ASYNC
  int4 st[6];   // fallback staging registers (6 x 16B per thread per stage)
#endif

  // A: 512 16B-chunks per array (thread -> chunks 2t, 2t+1);
  // W: 256 16B-chunks per array (thread -> chunk t).
  auto fetchStage = [&](int kt, int buf) {
    const int k0 = kt * BK;
#if QC_ASYNC
#pragma unroll
    for (int c = 0; c < 2; ++c) {
      const int chunk = tid * 2 + c;
      const int row = chunk >> 2, off = (chunk & 3) * 8;
      copy16_async(&Ahi[(size_t)(m0 + row) * QDIM + k0 + off],
                   &sAhi[buf][row * LDSK + off]);
      copy16_async(&Alo[(size_t)(m0 + row) * QDIM + k0 + off],
                   &sAlo[buf][row * LDSK + off]);
    }
    {
      const int row = tid >> 2, off = (tid & 3) * 8;
      copy16_async(&Wthi[(size_t)(n0 + row) * QDIM + k0 + off],
                   &sWhi[buf][row * LDSK + off]);
      copy16_async(&Wtlo[(size_t)(n0 + row) * QDIM + k0 + off],
                   &sWlo[buf][row * LDSK + off]);
    }
#else
#pragma unroll
    for (int c = 0; c < 2; ++c) {
      const int chunk = tid * 2 + c;
      const int row = chunk >> 2, off = (chunk & 3) * 8;
      st[c]     = *reinterpret_cast<const int4*>(&Ahi[(size_t)(m0 + row) * QDIM + k0 + off]);
      st[2 + c] = *reinterpret_cast<const int4*>(&Alo[(size_t)(m0 + row) * QDIM + k0 + off]);
    }
    {
      const int row = tid >> 2, off = (tid & 3) * 8;
      st[4] = *reinterpret_cast<const int4*>(&Wthi[(size_t)(n0 + row) * QDIM + k0 + off]);
      st[5] = *reinterpret_cast<const int4*>(&Wtlo[(size_t)(n0 + row) * QDIM + k0 + off]);
    }
#endif
  };

  auto publishStage = [&](int buf) {
#if !QC_ASYNC
#pragma unroll
    for (int c = 0; c < 2; ++c) {
      const int chunk = tid * 2 + c;
      const int row = chunk >> 2, off = (chunk & 3) * 8;
      *reinterpret_cast<int4*>(&sAhi[buf][row * LDSK + off]) = st[c];
      *reinterpret_cast<int4*>(&sAlo[buf][row * LDSK + off]) = st[2 + c];
    }
    {
      const int row = tid >> 2, off = (tid & 3) * 8;
      *reinterpret_cast<int4*>(&sWhi[buf][row * LDSK + off]) = st[4];
      *reinterpret_cast<int4*>(&sWlo[buf][row * LDSK + off]) = st[5];
    }
#else
    (void)buf;
#endif
  };

  // ---- compute mapping: 8 waves = 4(M) x 2(N); wave tile = 32 rows x 32 cols ----
  const int lane    = tid & 31;
  const int wave    = tid >> 5;
  const int waveM   = wave & 3;
  const int waveN   = wave >> 2;
  const int lr      = lane & 15;
  const bool hiHalf = (lane >= 16);

  v8f acc[2][2];
#pragma unroll
  for (int mt = 0; mt < 2; ++mt)
#pragma unroll
    for (int nt = 0; nt < 2; ++nt)
#pragma unroll
      for (int i = 0; i < 8; ++i) acc[mt][nt][i] = 0.f;

  auto computeTile = [&](int buf) {
    const int kA = hiHalf ? 8 : 0;    // A frag K runs {kA..kA+7} u {kA+16..kA+23}
    v16bf ahi[2], alo[2];
#pragma unroll
    for (int mt = 0; mt < 2; ++mt) {
      const int row = waveM * 32 + mt * 16 + lr;
      v8bf h0 = *reinterpret_cast<const v8bf*>(&sAhi[buf][row * LDSK + kA]);
      v8bf h1 = *reinterpret_cast<const v8bf*>(&sAhi[buf][row * LDSK + kA + 16]);
      ahi[mt] = __builtin_shufflevector(h0, h1, 0, 1, 2, 3, 4, 5, 6, 7,
                                        8, 9, 10, 11, 12, 13, 14, 15);
      v8bf l0 = *reinterpret_cast<const v8bf*>(&sAlo[buf][row * LDSK + kA]);
      v8bf l1 = *reinterpret_cast<const v8bf*>(&sAlo[buf][row * LDSK + kA + 16]);
      alo[mt] = __builtin_shufflevector(l0, l1, 0, 1, 2, 3, 4, 5, 6, 7,
                                        8, 9, 10, 11, 12, 13, 14, 15);
    }
    const int kB = hiHalf ? 8 : 0;    // B frag K runs {kB..kB+7} u {kB+16..kB+23}
#pragma unroll
    for (int nt = 0; nt < 2; ++nt) {
      const int col = waveN * 32 + nt * 16 + lr;
      v8bf h0 = *reinterpret_cast<const v8bf*>(&sWhi[buf][col * LDSK + kB]);
      v8bf h1 = *reinterpret_cast<const v8bf*>(&sWhi[buf][col * LDSK + kB + 16]);
      v16bf bhi = __builtin_shufflevector(h0, h1, 0, 1, 2, 3, 4, 5, 6, 7,
                                          8, 9, 10, 11, 12, 13, 14, 15);
      v8bf l0 = *reinterpret_cast<const v8bf*>(&sWlo[buf][col * LDSK + kB]);
      v8bf l1 = *reinterpret_cast<const v8bf*>(&sWlo[buf][col * LDSK + kB + 16]);
      v16bf blo = __builtin_shufflevector(l0, l1, 0, 1, 2, 3, 4, 5, 6, 7,
                                          8, 9, 10, 11, 12, 13, 14, 15);
#pragma unroll
      for (int mt = 0; mt < 2; ++mt) {
        acc[mt][nt] = __builtin_amdgcn_wmma_f32_16x16x32_bf16(
            false, ahi[mt], false, bhi, (short)0, acc[mt][nt], false, false);
        acc[mt][nt] = __builtin_amdgcn_wmma_f32_16x16x32_bf16(
            false, ahi[mt], false, blo, (short)0, acc[mt][nt], false, false);
        acc[mt][nt] = __builtin_amdgcn_wmma_f32_16x16x32_bf16(
            false, alo[mt], false, bhi, (short)0, acc[mt][nt], false, false);
      }
    }
  };

  // ---- pipelined main loop: one barrier per iteration, copies overlap compute ----
  fetchStage(0, 0);
  publishStage(0);
  for (int kt = 0; kt < KTILES; ++kt) {
    const int cur = kt & 1;
    qc_wait_async();               // async copies for buf[cur] have landed
    __syncthreads();               // tile visible; everyone done with buf[cur^1]
    if (kt + 1 < KTILES) fetchStage(kt + 1, cur ^ 1);
    computeTile(cur);
    if (kt + 1 < KTILES) publishStage(cur ^ 1);
  }

  // ---- epilogue: C/D layout — VGPR r -> row r (+8 for lanes 16..31), col = lane&15
#pragma unroll
  for (int mt = 0; mt < 2; ++mt) {
    const int gmBase = m0 + waveM * 32 + mt * 16 + (hiHalf ? 8 : 0);
#pragma unroll
    for (int nt = 0; nt < 2; ++nt) {
      const int gn = n0 + waveN * 32 + nt * 16 + lr;
      if (Cf32) {
#pragma unroll
        for (int r = 0; r < 8; ++r)
          Cf32[(size_t)(gmBase + r) * QDIM + gn] = acc[mt][nt][r];
      } else {
        const float s = nextScale ? nextScale[gn] : 1.0f;
#pragma unroll
        for (int r = 0; r < 8; ++r) {
          __bf16 h, l;
          split_bf16(acc[mt][nt][r] * s, h, l);
          Chi[(size_t)(gmBase + r) * QDIM + gn] = h;
          Clo[(size_t)(gmBase + r) * QDIM + gn] = l;
        }
      }
    }
  }
}

extern "C" void kernel_launch(void* const* d_in, const int* in_sizes, int n_in,
                              void* d_out, int out_size, void* d_ws, size_t ws_size,
                              hipStream_t stream) {
  (void)in_sizes; (void)n_in; (void)out_size; (void)ws_size;

  const float* input   = (const float*)d_in[0];  // [4096, 2048]
  const float* angles  = (const float*)d_in[1];  // [16, 2048, 3]
  const float* weights = (const float*)d_in[2];  // [16, 2048, 2048]
  const float* basis   = (const float*)d_in[3];  // [2048, 2048]
  float* out = (float*)d_out;                    // [4096, 2048]

  // workspace layout (~80.1 MB):
  //   scales (128 KB) | Wt_hi (8 MB) | Wt_lo (8 MB) | 4 state bufs (16 MB each)
  char* ws = (char*)d_ws;
  float*  scales = (float*)ws;                       ws += (size_t)NLAYERS * QDIM * 4;
  __bf16* wthi   = (__bf16*)ws;                      ws += (size_t)QDIM * QDIM * 2;
  __bf16* wtlo   = (__bf16*)ws;                      ws += (size_t)QDIM * QDIM * 2;
  __bf16* sbuf[4];
  for (int i = 0; i < 4; ++i) { sbuf[i] = (__bf16*)ws; ws += (size_t)BATCH * QDIM * 2; }
  // ping-pong pairs: state l lives in {sbuf[2*(l&1)], sbuf[2*(l&1)+1]} (hi, lo)

  {
    int n = NLAYERS * QDIM;
    qc_scales_kernel<<<(n + 255) / 256, 256, 0, stream>>>(angles, scales, n);
  }
  {
    size_t n = (size_t)BATCH * QDIM;
    qc_split_in<<<(unsigned)((n + 255) / 256), 256, 0, stream>>>(
        input, scales /* layer 0 */, sbuf[0], sbuf[1]);
  }

  dim3 tgrid(QDIM / 32, QDIM / 32);   // (64, 64) for transpose/split
  dim3 ggrid(QDIM / BN, BATCH / BM);  // (32, 32) for GEMM
  dim3 block(256);

  for (int l = 0; l < NLAYERS; ++l) {
    qc_split_w<<<tgrid, block, 0, stream>>>(weights + (size_t)l * QDIM * QDIM, wthi, wtlo);
    __bf16* inHi  = sbuf[2 * (l & 1) + 0];
    __bf16* inLo  = sbuf[2 * (l & 1) + 1];
    __bf16* outHi = sbuf[2 * ((l + 1) & 1) + 0];
    __bf16* outLo = sbuf[2 * ((l + 1) & 1) + 1];
    const float* nsc = (l + 1 < NLAYERS) ? scales + (size_t)(l + 1) * QDIM : nullptr;
    qc_gemm_bf16<<<ggrid, block, 0, stream>>>(inHi, inLo, wthi, wtlo, nsc,
                                              outHi, outLo, nullptr);
  }
  // final measurement GEMM -> fp32 output
  qc_split_w<<<tgrid, block, 0, stream>>>(basis, wthi, wtlo);
  qc_gemm_bf16<<<ggrid, block, 0, stream>>>(sbuf[0], sbuf[1], wthi, wtlo, nullptr,
                                            nullptr, nullptr, out);
}
